// MultiHeadedSelfAttention_12730283065399
// MI455X (gfx1250) — matrix operand order repaired and checked by reading, not compiled, not run on hardware
//
#include <hip/hip_runtime.h>

typedef __attribute__((ext_vector_type(8)))  __bf16 v8bf;
typedef __attribute__((ext_vector_type(16))) __bf16 v16bf;
typedef __attribute__((ext_vector_type(8)))  float  v8f;

#define D_DIM 1024
#define S_LEN 1024
#define BATCH 8
#define HEADS 16
#define HW    64

static __device__ inline v16bf make_frag(v8bf lo, v8bf hi) {
  v16bf r;
#pragma unroll
  for (int i = 0; i < 8; ++i) { r[i] = lo[i]; r[8 + i] = hi[i]; }
  return r;
}

// ---------------- packing kernels ----------------
__global__ void k_f32_to_bf16(const float* __restrict__ in, __bf16* __restrict__ out, int n) {
  int i = blockIdx.x * blockDim.x + threadIdx.x;
  if (i < n) out[i] = (__bf16)in[i];
}

// W (D,D) fp32 row-major -> Wt bf16 with Wt[n*D+k] = W[k*D+n]
__global__ void k_pack_wt(const float* __restrict__ W, __bf16* __restrict__ Wt) {
  int i = blockIdx.x * blockDim.x + threadIdx.x;
  int k = i >> 10;
  int n = i & (D_DIM - 1);
  Wt[(size_t)n * D_DIM + k] = (__bf16)W[i];
}

// ---------------- GEMM: out[m,n] = sum_k A[m,k] * W[k,n] + bias[n] ----------------
// Each wave: 64x64 macro-tile (4x4 WMMA grid). All 8 waves of a block share the same
// 64 A-rows, so the 64x32 A tile is staged into LDS once per K-step with
// global_load_async_to_lds_b128 (each of 256 lanes copies 16B), then read as A-fragments.
// vmode 0: store bf16 row-major out[m*D+n].
// vmode 1: store bf16 V-transposed: out[(b*D+n)*S + s]  (m = b*S + s)
__global__ void __launch_bounds__(256, 1)
k_gemm_bf16(const __bf16* __restrict__ A, const __bf16* __restrict__ Wt,
            const float* __restrict__ bias, __bf16* __restrict__ out, int vmode) {
  const int ln = threadIdx.x & 31, lo = ln & 15, hi = ln >> 4;
  const int wave = blockIdx.x * (blockDim.x >> 5) + (threadIdx.x >> 5);
  const int nt = wave & 15;         // 16 N-tiles of 64
  const int mt = wave >> 4;         // 128 M-tiles of 64 (uniform within a block)
  const int m0 = mt * 64, n0 = nt * 64;

  __shared__ __bf16 atile[64][32];  // 4 KB staged A tile, shared by all 8 waves

  // async-copy assignment: thread t copies 16B at (row = t/4, col = (t%4)*8 elems)
  const int tid  = threadIdx.x;
  const int srow = tid >> 2;
  const int scol = (tid & 3) << 3;
  const unsigned lds_dst = (unsigned)(size_t)&atile[srow][scol];
  const __bf16* gsrc = A + (size_t)(m0 + srow) * D_DIM + scol;

  v8f acc[4][4] = {};
  size_t brow[4];
#pragma unroll
  for (int j = 0; j < 4; ++j) brow[j] = (size_t)(n0 + 16 * j + lo) * D_DIM;

#pragma unroll 1
  for (int kb = 0; kb < D_DIM; kb += 32) {
    // stage A tile for this K-step (async DMA into LDS)
    asm volatile("global_load_async_to_lds_b128 %0, %1, off"
                 :: "v"(lds_dst), "v"((unsigned long long)(size_t)(gsrc + kb))
                 : "memory");
    asm volatile("s_wait_asynccnt 0" ::: "memory");  // own copies landed
    __syncthreads();                                 // all waves' copies visible
    __builtin_prefetch((const void*)(Wt + brow[0] + kb + 64), 0, 0);

    v16bf af[4], bf[4];
#pragma unroll
    for (int i = 0; i < 4; ++i) {
      const __bf16* ap = &atile[16 * i + lo][hi * 16];
      af[i] = make_frag(*(const v8bf*)ap, *(const v8bf*)(ap + 8));
    }
#pragma unroll
    for (int j = 0; j < 4; ++j)
      bf[j] = *(const v16bf*)(Wt + brow[j] + kb + hi * 16);
#pragma unroll
    for (int i = 0; i < 4; ++i)
#pragma unroll
      for (int j = 0; j < 4; ++j)
        acc[i][j] = __builtin_amdgcn_wmma_f32_16x16x32_bf16(false, af[i], false, bf[j],
                                                            (short)0, acc[i][j], false, false);
    __syncthreads();                                 // reads done before next overwrite
  }

#pragma unroll
  for (int j = 0; j < 4; ++j) {
    const float bv = bias[n0 + 16 * j + lo];
    if (vmode == 0) {
#pragma unroll
      for (int i = 0; i < 4; ++i)
#pragma unroll
        for (int r = 0; r < 8; ++r) {
          int m = m0 + 16 * i + r + 8 * hi;
          out[(size_t)m * D_DIM + n0 + 16 * j + lo] = (__bf16)(acc[i][j][r] + bv);
        }
    } else {
#pragma unroll
      for (int i = 0; i < 4; ++i)
#pragma unroll
        for (int r = 0; r < 8; ++r) {
          int m = m0 + 16 * i + r + 8 * hi;
          int bb = m >> 10, s = m & (S_LEN - 1);
          out[((size_t)bb * D_DIM + n0 + 16 * j + lo) * S_LEN + s] = (__bf16)(acc[i][j][r] + bv);
        }
    }
  }
}

// Same register-blocked GEMM with LDS A-staging, fp32 output (final projection)
__global__ void __launch_bounds__(256, 1)
k_gemm_out(const __bf16* __restrict__ A, const __bf16* __restrict__ Wt,
           const float* __restrict__ bias, float* __restrict__ out) {
  const int ln = threadIdx.x & 31, lo = ln & 15, hi = ln >> 4;
  const int wave = blockIdx.x * (blockDim.x >> 5) + (threadIdx.x >> 5);
  const int nt = wave & 15;
  const int mt = wave >> 4;
  const int m0 = mt * 64, n0 = nt * 64;

  __shared__ __bf16 atile[64][32];
  const int tid  = threadIdx.x;
  const int srow = tid >> 2;
  const int scol = (tid & 3) << 3;
  const unsigned lds_dst = (unsigned)(size_t)&atile[srow][scol];
  const __bf16* gsrc = A + (size_t)(m0 + srow) * D_DIM + scol;

  v8f acc[4][4] = {};
  size_t brow[4];
#pragma unroll
  for (int j = 0; j < 4; ++j) brow[j] = (size_t)(n0 + 16 * j + lo) * D_DIM;

#pragma unroll 1
  for (int kb = 0; kb < D_DIM; kb += 32) {
    asm volatile("global_load_async_to_lds_b128 %0, %1, off"
                 :: "v"(lds_dst), "v"((unsigned long long)(size_t)(gsrc + kb))
                 : "memory");
    asm volatile("s_wait_asynccnt 0" ::: "memory");
    __syncthreads();
    __builtin_prefetch((const void*)(Wt + brow[0] + kb + 64), 0, 0);

    v16bf af[4], bf[4];
#pragma unroll
    for (int i = 0; i < 4; ++i) {
      const __bf16* ap = &atile[16 * i + lo][hi * 16];
      af[i] = make_frag(*(const v8bf*)ap, *(const v8bf*)(ap + 8));
    }
#pragma unroll
    for (int j = 0; j < 4; ++j)
      bf[j] = *(const v16bf*)(Wt + brow[j] + kb + hi * 16);
#pragma unroll
    for (int i = 0; i < 4; ++i)
#pragma unroll
      for (int j = 0; j < 4; ++j)
        acc[i][j] = __builtin_amdgcn_wmma_f32_16x16x32_bf16(false, af[i], false, bf[j],
                                                            (short)0, acc[i][j], false, false);
    __syncthreads();
  }

#pragma unroll
  for (int j = 0; j < 4; ++j) {
    const float bv = bias[n0 + 16 * j + lo];
#pragma unroll
    for (int i = 0; i < 4; ++i)
#pragma unroll
      for (int r = 0; r < 8; ++r) {
        int m = m0 + 16 * i + r + 8 * hi;
        out[(size_t)m * D_DIM + n0 + 16 * j + lo] = acc[i][j][r] + bv;
      }
  }
}

// ---------------- flash attention, one wave per (b, h, 32-query tile) ----------------
// Q,K: (B*S, D) bf16 row-major. Vt: (B, D, S) bf16. ctx: (B*S, D) bf16 row-major.
// Wave-local only (divergent trip counts -> no block barriers).
__global__ void __launch_bounds__(256, 1)
k_attn(const __bf16* __restrict__ Q, const __bf16* __restrict__ K,
       const __bf16* __restrict__ Vt, __bf16* __restrict__ ctx) {
  const int ln = threadIdx.x & 31, lo = ln & 15, hi = ln >> 4;
  const int wid = threadIdx.x >> 5;
  const int wave = blockIdx.x * (blockDim.x >> 5) + wid;
  const int qt = wave & 31;          // S/32 = 32 query tiles
  const int h  = (wave >> 5) & 15;
  const int bb = wave >> 9;
  const int q0 = qt * 32;

  __shared__ __bf16 lds[8][32][32];  // per-wave P tile (32 queries x 32 keys)

  // Q A-fragments: two 16-row M-frags, each split into two K=32 chunks
  v16bf qf[2][2];
#pragma unroll
  for (int mi = 0; mi < 2; ++mi) {
    const __bf16* qp = Q + (size_t)(bb * S_LEN + q0 + 16 * mi + lo) * D_DIM + h * HW + hi * 16;
    qf[mi][0] = make_frag(*(const v8bf*)qp,        *(const v8bf*)(qp + 8));
    qf[mi][1] = make_frag(*(const v8bf*)(qp + 32), *(const v8bf*)(qp + 40));
  }

  v8f acc[2][4] = {};
  float mrow[2][8], lrow[2][8];
#pragma unroll
  for (int mi = 0; mi < 2; ++mi)
#pragma unroll
    for (int r = 0; r < 8; ++r) { mrow[mi][r] = -1e30f; lrow[mi][r] = 0.0f; }

  const int ktmax = (q0 + 31) >> 5;
#pragma unroll 1
  for (int kt = 0; kt <= ktmax; ++kt) {
    const int k0 = kt * 32;
    // ---- scores S = Q K^T (two 16-key halves x two M-frags) ----
    v8f s[2][2] = {};
#pragma unroll
    for (int nh = 0; nh < 2; ++nh) {
      const __bf16* kp = K + (size_t)(bb * S_LEN + k0 + nh * 16 + lo) * D_DIM + h * HW;
      v16bf b0 = *(const v16bf*)(kp + hi * 16);        // head dims 0..31
      v16bf b1 = *(const v16bf*)(kp + 32 + hi * 16);   // head dims 32..63
#pragma unroll
      for (int mi = 0; mi < 2; ++mi) {
        s[mi][nh] = __builtin_amdgcn_wmma_f32_16x16x32_bf16(false, qf[mi][0], false, b0,
                                                            (short)0, s[mi][nh], false, false);
        s[mi][nh] = __builtin_amdgcn_wmma_f32_16x16x32_bf16(false, qf[mi][1], false, b1,
                                                            (short)0, s[mi][nh], false, false);
      }
    }
    // ---- scale + causal mask + online softmax ----
#pragma unroll
    for (int mi = 0; mi < 2; ++mi) {
#pragma unroll
      for (int r = 0; r < 8; ++r) {
        const int qrow = q0 + 16 * mi + r + 8 * hi;
        float v0 = s[mi][0][r] * 0.125f;   // 1/sqrt(64)
        float v1 = s[mi][1][r] * 0.125f;
        if (k0 + lo      > qrow) v0 = -1e30f;
        if (k0 + 16 + lo > qrow) v1 = -1e30f;
        float cmax = fmaxf(v0, v1);
#pragma unroll
        for (int msk = 1; msk < 16; msk <<= 1)
          cmax = fmaxf(cmax, __shfl_xor(cmax, msk, 32));
        const float mnew  = fmaxf(mrow[mi][r], cmax);
        const float alpha = __expf(mrow[mi][r] - mnew);
        const float p0 = __expf(v0 - mnew);
        const float p1 = __expf(v1 - mnew);
        float ps = p0 + p1;
#pragma unroll
        for (int msk = 1; msk < 16; msk <<= 1)
          ps += __shfl_xor(ps, msk, 32);
        lrow[mi][r] = lrow[mi][r] * alpha + ps;
        mrow[mi][r] = mnew;
        s[mi][0][r] = p0; s[mi][1][r] = p1;
        acc[mi][0][r] *= alpha; acc[mi][1][r] *= alpha;
        acc[mi][2][r] *= alpha; acc[mi][3][r] *= alpha;
      }
    }
    // ---- P: C-layout -> LDS row-major -> reload as A-fragments (wave-local transpose) ----
#pragma unroll
    for (int mi = 0; mi < 2; ++mi)
#pragma unroll
      for (int r = 0; r < 8; ++r) {
        lds[wid][16 * mi + r + 8 * hi][lo]      = (__bf16)s[mi][0][r];
        lds[wid][16 * mi + r + 8 * hi][16 + lo] = (__bf16)s[mi][1][r];
      }
    asm volatile("s_wait_dscnt 0" ::: "memory");  // wave-local DS RAW fence
    v16bf pf[2];
#pragma unroll
    for (int mi = 0; mi < 2; ++mi) {
      const __bf16* pp = &lds[wid][16 * mi + lo][hi * 16];
      pf[mi] = make_frag(*(const v8bf*)pp, *(const v8bf*)(pp + 8));
    }
    // ---- O += P (32x32) * V (32x64): shared V B-fragments across both M-frags ----
    const __bf16* vbase = Vt + ((size_t)bb * D_DIM + h * HW) * S_LEN;
#pragma unroll
    for (int j = 0; j < 4; ++j) {
      v16bf vb = *(const v16bf*)(vbase + (size_t)(16 * j + lo) * S_LEN + k0 + hi * 16);
#pragma unroll
      for (int mi = 0; mi < 2; ++mi)
        acc[mi][j] = __builtin_amdgcn_wmma_f32_16x16x32_bf16(false, pf[mi], false, vb,
                                                             (short)0, acc[mi][j], false, false);
    }
  }

  // ---- normalize + store ctx (row-major bf16) ----
#pragma unroll
  for (int mi = 0; mi < 2; ++mi)
#pragma unroll
    for (int r = 0; r < 8; ++r) {
      const float inv = 1.0f / lrow[mi][r];
      __bf16* cp = ctx + (size_t)(bb * S_LEN + q0 + 16 * mi + r + 8 * hi) * D_DIM + h * HW + lo;
      cp[0]  = (__bf16)(acc[mi][0][r] * inv);
      cp[16] = (__bf16)(acc[mi][1][r] * inv);
      cp[32] = (__bf16)(acc[mi][2][r] * inv);
      cp[48] = (__bf16)(acc[mi][3][r] * inv);
    }
}

extern "C" void kernel_launch(void* const* d_in, const int* in_sizes, int n_in,
                              void* d_out, int out_size, void* d_ws, size_t ws_size,
                              hipStream_t stream) {
  const float* x  = (const float*)d_in[0];
  const float* Wq = (const float*)d_in[1];
  const float* bq = (const float*)d_in[2];
  const float* Wk = (const float*)d_in[3];
  const float* bk = (const float*)d_in[4];
  const float* Wv = (const float*)d_in[5];
  const float* bv = (const float*)d_in[6];
  const float* Wo = (const float*)d_in[7];
  const float* bo = (const float*)d_in[8];
  float* out = (float*)d_out;

  char* ws = (char*)d_ws;
  const size_t MB = 1ull << 20;
  __bf16* xb  = (__bf16*)(ws);             // 16 MB: x in bf16
  __bf16* Wqt = (__bf16*)(ws + 16 * MB);   // 2 MB each: transposed bf16 weights
  __bf16* Wkt = (__bf16*)(ws + 18 * MB);
  __bf16* Wvt = (__bf16*)(ws + 20 * MB);
  __bf16* Wot = (__bf16*)(ws + 22 * MB);
  __bf16* Qb  = (__bf16*)(ws + 24 * MB);   // 16 MB
  __bf16* Kb  = (__bf16*)(ws + 40 * MB);   // 16 MB
  __bf16* Vtb = (__bf16*)(ws + 56 * MB);   // 16 MB (B, D, S)
  __bf16* Cb  = (__bf16*)(ws + 72 * MB);   // 16 MB attention context

  const int nx = BATCH * S_LEN * D_DIM;    // 8M
  k_f32_to_bf16<<<nx / 256, 256, 0, stream>>>(x, xb, nx);
  const int nw = D_DIM * D_DIM;            // 1M
  k_pack_wt<<<nw / 256, 256, 0, stream>>>(Wq, Wqt);
  k_pack_wt<<<nw / 256, 256, 0, stream>>>(Wk, Wkt);
  k_pack_wt<<<nw / 256, 256, 0, stream>>>(Wv, Wvt);
  k_pack_wt<<<nw / 256, 256, 0, stream>>>(Wo, Wot);

  // 128 M-tiles x 16 N-tiles = 2048 waves / 8 per block = 256 blocks
  k_gemm_bf16<<<256, 256, 0, stream>>>(xb, Wqt, bq, Qb, 0);
  k_gemm_bf16<<<256, 256, 0, stream>>>(xb, Wkt, bk, Kb, 0);
  k_gemm_bf16<<<256, 256, 0, stream>>>(xb, Wvt, bv, Vtb, 1);

  // B*H*(S/32) = 4096 waves / 8 per block = 512 blocks
  k_attn<<<512, 256, 0, stream>>>(Qb, Kb, Vtb, Cb);

  k_gemm_out<<<256, 256, 0, stream>>>(Cb, Wot, bo, out);
}
